// TransformerLayer_2869038154396
// MI455X (gfx1250) — compile-verified, matrix operand order checked by reading, not run on hardware
//
#include <hip/hip_runtime.h>
#include <hip/hip_bf16.h>
#include <math.h>

typedef _Float16 v16h __attribute__((ext_vector_type(16)));
typedef _Float16 v8h  __attribute__((ext_vector_type(8)));
typedef float    v8f  __attribute__((ext_vector_type(8)));
typedef int      vi4  __attribute__((__vector_size__(16)));

#define AS1 __attribute__((address_space(1)))
#define AS3 __attribute__((address_space(3)))

#define BATCH   8192
#define WIN     10
#define DMODEL  1024
#define NHEAD   4
#define KDIM    8
#define VDIM    8
#define QKVN    96              // 32 K-cols + 32 Q-cols + 32 V-cols
#define ROWS    (BATCH * WIN)   // 81920, divisible by 32

// ---- workspace byte offsets ----
#define OFF_WQKVT  ((size_t)0)                         // 96*1024 f16      = 196608 B
#define OFF_WLT    ((size_t)196608)                    // 1024*32 f16      = 65536 B
#define OFF_W2T    ((size_t)262144)                    // 1024*1024 f16    = 2 MiB
#define OFF_W3T    ((size_t)2359296)                   // 1024*1024 f16    = 2 MiB
#define OFF_Y32    ((size_t)4456448)                   // 81920*1024 f32   = 320 MiB
#define OFF_Y16    ((size_t)340000768)                 // 81920*1024 f16   = 160 MiB
#define OFF_H16    ((size_t)507772928)                 // 81920*1024 f16   = 160 MiB

#if __has_builtin(__builtin_amdgcn_global_load_async_to_lds_b128)
#define HAS_ASYNC_LDS 1
#endif

__device__ inline void wait_async0() {
#if __has_builtin(__builtin_amdgcn_s_wait_asynccnt)
    __builtin_amdgcn_s_wait_asynccnt(0);
#else
    asm volatile("s_wait_asynccnt 0" ::: "memory");
#endif
}

// Copy `n16` 16-byte chunks global -> LDS, 512 threads, async path if available.
__device__ inline void stage_tile(const uint4* __restrict__ src, uint4* dst, int n16, int t) {
#ifdef HAS_ASYNC_LDS
    for (int p = t; p < n16; p += 512)
        __builtin_amdgcn_global_load_async_to_lds_b128(
            (AS1 vi4*)(src + p), (AS3 vi4*)(dst + p), 0, 0);
    wait_async0();
#else
    for (int p = t; p < n16; p += 512) dst[p] = src[p];
#endif
}

// Build a 16x32 f16 A/B fragment per the CDNA5 16-bit VGPR layout:
// lane L<16  : halves [0..7]=K(k0..k0+7)    [8..15]=K(k0+16..k0+23)
// lane L>=16 : halves [0..7]=K(k0+8..15)    [8..15]=K(k0+24..k0+31)
// Caller passes p already offset by (hi ? 8 : 0) halves.
__device__ inline v16h ldfrag(const _Float16* p) {
    v8h lo = *(const v8h*)(p);
    v8h hh = *(const v8h*)(p + 16);
    v16h r;
#pragma unroll
    for (int i = 0; i < 8; ++i) { r[i] = lo[i]; r[i + 8] = hh[i]; }
    return r;
}

__device__ inline v8f wmma_f16(v16h a, v16h b, v8f c) {
    return __builtin_amdgcn_wmma_f32_16x16x32_f16(false, a, false, b, (short)0, c, false, false);
}

// ---------------- prep kernels: transpose + fp32 -> f16 ----------------
__global__ void prep_qkvT(const float* __restrict__ wq, const float* __restrict__ wk,
                          const float* __restrict__ wv, _Float16* __restrict__ out) {
    int idx = blockIdx.x * blockDim.x + threadIdx.x;
    if (idx >= QKVN * DMODEL) return;
    int n = idx / DMODEL, d = idx % DMODEL;
    int sel = n >> 5, rem = n & 31, h = rem >> 3, j = rem & 7;
    const float* w = (sel == 0) ? wk : (sel == 1 ? wq : wv);   // cols: [K | Q | V]
    out[(size_t)n * DMODEL + d] = (_Float16)w[((size_t)h * DMODEL + d) * 8 + j];
}

__global__ void prep_T(const float* __restrict__ src, _Float16* __restrict__ dst,
                       int rows, int cols) {
    int idx = blockIdx.x * blockDim.x + threadIdx.x;
    if (idx >= rows * cols) return;
    int c = idx / rows, r = idx % rows;                 // dst[c][r] = src[r][c]
    dst[(size_t)c * rows + r] = (_Float16)src[(size_t)r * cols + c];
}

// ---------------- attention + Wl + residual + l2norm ----------------
__global__ __launch_bounds__(128) void attn_kernel(
    const float* __restrict__ x, const _Float16* __restrict__ wqkvT,
    const _Float16* __restrict__ wlT, const float* __restrict__ bl,
    float* __restrict__ y32, _Float16* __restrict__ y16) {

    __shared__ __align__(16) char ubuf[WIN * DMODEL * 4];    // 40 KiB: xs(f16) then yls(f32)
    __shared__ __align__(16) float qkv[16 * QKVN];           // 6 KiB
    __shared__ float sc[NHEAD * WIN * WIN];
    __shared__ __align__(16) _Float16 ao[16 * 32];
    __shared__ float ps[WIN * 8];
    __shared__ float rscale[WIN];

    _Float16* xs = (_Float16*)ubuf;   // 16x1024 f16 (rows 10..15 zero)
    float*   yls = (float*)ubuf;      // 10x1024 f32 (reused after QKV)

    const int b = blockIdx.x, t = threadIdx.x;
    const int wave = t >> 5, lane = t & 31, rL = lane & 15, hi = lane >> 4;
    const float* xb = x + (size_t)b * WIN * DMODEL;

    // stage x_b as f16 (pad rows 10..15 with zeros)
    for (int p = t * 4; p < 16 * DMODEL; p += 128 * 4) {
        int r = p >> 10, c = p & 1023;
        if (r < WIN) {
            float4 v = *(const float4*)&xb[r * DMODEL + c];
            xs[p + 0] = (_Float16)v.x; xs[p + 1] = (_Float16)v.y;
            xs[p + 2] = (_Float16)v.z; xs[p + 3] = (_Float16)v.w;
        } else {
            xs[p + 0] = (_Float16)0.f; xs[p + 1] = (_Float16)0.f;
            xs[p + 2] = (_Float16)0.f; xs[p + 3] = (_Float16)0.f;
        }
    }
    __syncthreads();

    // QKV = x16 @ WqkvT^T : M=16, N=96 (6 tiles), K=1024
    const _Float16* arow = xs + rL * DMODEL + hi * 8;
    for (int tile = wave; tile < 6; tile += 4) {
        const _Float16* brow = wqkvT + (size_t)(tile * 16 + rL) * DMODEL + hi * 8;
        v8f acc = {};
        for (int k0 = 0; k0 < DMODEL; k0 += 32)
            acc = wmma_f16(ldfrag(arow + k0), ldfrag(brow + k0), acc);
#pragma unroll
        for (int r = 0; r < 8; ++r)
            qkv[(r + hi * 8) * QKVN + tile * 16 + rL] = acc[r];
    }
    __syncthreads();

    // scores: a[i][j] = k_i . q_j, masked via min with tri
    if (wave < NHEAD) {
        for (int p = lane; p < WIN * WIN; p += 32) {
            int i = p / WIN, j = p - i * WIN;
            float s = 0.f;
#pragma unroll
            for (int k = 0; k < KDIM; ++k)
                s += qkv[i * QKVN + wave * 8 + k] * qkv[j * QKVN + 32 + wave * 8 + k];
            float tri = (i >= j) ? 1e5f : -1e5f;
            sc[(wave * WIN + i) * WIN + j] = fminf(s, tri);
        }
    }
    __syncthreads();

    // softmax over j (40 independent rows)
    if (t < NHEAD * WIN) {
        float* row = &sc[t * WIN];
        float m = row[0];
#pragma unroll
        for (int j = 1; j < WIN; ++j) m = fmaxf(m, row[j]);
        float s = 0.f;
#pragma unroll
        for (int j = 0; j < WIN; ++j) { row[j] = __expf(row[j] - m); s += row[j]; }
        float inv = 1.f / s;
#pragma unroll
        for (int j = 0; j < WIN; ++j) row[j] *= inv;
    }
    __syncthreads();

    // attn @ V  -> ao (16x32 f16, head-major feature order, padded rows zero)
    for (int p = t; p < 16 * 32; p += 128) {
        int i = p >> 5, n = p & 31;
        _Float16 o = (_Float16)0.f;
        if (i < WIN) {
            int h = n >> 3, v = n & 7;
            float s = 0.f;
#pragma unroll
            for (int j = 0; j < WIN; ++j)
                s += sc[(h * WIN + i) * WIN + j] * qkv[j * QKVN + 64 + h * 8 + v];
            o = (_Float16)s;
        }
        ao[p] = o;
    }
    __syncthreads();

    // y = x + ao @ Wl + bl : M=16, K=32 (one WMMA per tile), N=1024 (64 tiles)
    v16h aW = ldfrag(ao + rL * 32 + hi * 8);
    for (int tile = wave; tile < 64; tile += 4) {
        int cb = tile * 16;
        v16h bW = ldfrag(wlT + (size_t)(cb + rL) * 32 + hi * 8);
        v8f acc = {};
        acc = wmma_f16(aW, bW, acc);
#pragma unroll
        for (int r = 0; r < 8; ++r) {
            int row = r + hi * 8;
            if (row < WIN) {
                int c = cb + rL;
                yls[row * DMODEL + c] = xb[row * DMODEL + c] + acc[r] + bl[c];
            }
        }
    }
    __syncthreads();

    // per-row l2 norm
    if (t < WIN * 8) {
        int r = t >> 3, s0 = t & 7;
        float a = 0.f;
        for (int c = s0; c < DMODEL; c += 8) { float v = yls[r * DMODEL + c]; a += v * v; }
        ps[t] = a;
    }
    __syncthreads();
    if (t < WIN) {
        float ss = 0.f;
#pragma unroll
        for (int k = 0; k < 8; ++k) ss += ps[t * 8 + k];
        rscale[t] = 1.f / fmaxf(sqrtf(ss), 1e-12f);
    }
    __syncthreads();

    for (int p = t; p < WIN * DMODEL; p += 128) {
        int r = p >> 10;
        float v = yls[p] * rscale[r];
        size_t gi = (size_t)(b * WIN + r) * DMODEL + (p & 1023);
        y32[gi] = v;
        y16[gi] = (_Float16)v;
    }
}

// ---------------- FFN1: h = relu(y @ W2 + b2) -> f16 ----------------
// Workgroup: 512 threads (16 waves), 32 rows x 1024 cols; wave tile 32x64 (2Mx4N frags).
__global__ __launch_bounds__(512) void ffn1_kernel(
    const _Float16* __restrict__ y16, const _Float16* __restrict__ w2T,
    const float* __restrict__ b2, _Float16* __restrict__ h16) {

    __shared__ __align__(16) _Float16 as[32 * DMODEL];   // 64 KiB A tile
    const int t = threadIdx.x, wave = t >> 5, lane = t & 31, rL = lane & 15, hi = lane >> 4;
    const int m0 = blockIdx.x * 32;

    stage_tile((const uint4*)(y16 + (size_t)m0 * DMODEL), (uint4*)as, 32 * DMODEL / 8, t);
    __syncthreads();

    const int cb = wave * 64;
    const _Float16* arow[2];
    arow[0] = as + rL * DMODEL + hi * 8;
    arow[1] = as + (16 + rL) * DMODEL + hi * 8;
    const _Float16* brow[4];
#pragma unroll
    for (int tt = 0; tt < 4; ++tt)
        brow[tt] = w2T + (size_t)(cb + tt * 16 + rL) * DMODEL + hi * 8;

    v8f acc[2][4] = {};
    for (int k0 = 0; k0 < DMODEL; k0 += 32) {
        v16h a0 = ldfrag(arow[0] + k0);
        v16h a1 = ldfrag(arow[1] + k0);
#pragma unroll
        for (int tt = 0; tt < 4; ++tt) {
            v16h bf = ldfrag(brow[tt] + k0);
            acc[0][tt] = wmma_f16(a0, bf, acc[0][tt]);
            acc[1][tt] = wmma_f16(a1, bf, acc[1][tt]);
        }
    }

#pragma unroll
    for (int tt = 0; tt < 4; ++tt) {
        int col = cb + tt * 16 + rL;
        float bias = b2[col];
#pragma unroll
        for (int mf = 0; mf < 2; ++mf)
#pragma unroll
            for (int r = 0; r < 8; ++r) {
                int row = m0 + mf * 16 + r + hi * 8;
                float v = fmaxf(acc[mf][tt][r] + bias, 0.f);
                h16[(size_t)row * DMODEL + col] = (_Float16)v;
            }
    }
}

// ---------------- FFN2: out = l2norm(y + h @ W3 + b3) -> f32 ----------------
__global__ __launch_bounds__(512) void ffn2_kernel(
    const _Float16* __restrict__ h16, const _Float16* __restrict__ w3T,
    const float* __restrict__ b3, const float* __restrict__ y32,
    float* __restrict__ out) {

    __shared__ __align__(16) _Float16 as[32 * DMODEL];
    __shared__ float rowss[32];
    const int t = threadIdx.x, wave = t >> 5, lane = t & 31, rL = lane & 15, hi = lane >> 4;
    const int m0 = blockIdx.x * 32;

    if (t < 32) rowss[t] = 0.f;
    stage_tile((const uint4*)(h16 + (size_t)m0 * DMODEL), (uint4*)as, 32 * DMODEL / 8, t);
    __syncthreads();

    const int cb = wave * 64;
    const _Float16* arow[2];
    arow[0] = as + rL * DMODEL + hi * 8;
    arow[1] = as + (16 + rL) * DMODEL + hi * 8;
    const _Float16* brow[4];
#pragma unroll
    for (int tt = 0; tt < 4; ++tt)
        brow[tt] = w3T + (size_t)(cb + tt * 16 + rL) * DMODEL + hi * 8;

    v8f acc[2][4] = {};
    for (int k0 = 0; k0 < DMODEL; k0 += 32) {
        v16h a0 = ldfrag(arow[0] + k0);
        v16h a1 = ldfrag(arow[1] + k0);
#pragma unroll
        for (int tt = 0; tt < 4; ++tt) {
            v16h bf = ldfrag(brow[tt] + k0);
            acc[0][tt] = wmma_f16(a0, bf, acc[0][tt]);
            acc[1][tt] = wmma_f16(a1, bf, acc[1][tt]);
        }
    }

    // t = y + z + b3 ; accumulate per-row sum of squares
#pragma unroll
    for (int tt = 0; tt < 4; ++tt) {
        int col = cb + tt * 16 + rL;
        float bias = b3[col];
#pragma unroll
        for (int mf = 0; mf < 2; ++mf)
#pragma unroll
            for (int r = 0; r < 8; ++r) {
                int row = m0 + mf * 16 + r + hi * 8;
                acc[mf][tt][r] = y32[(size_t)row * DMODEL + col] + acc[mf][tt][r] + bias;
            }
    }
#pragma unroll
    for (int mf = 0; mf < 2; ++mf)
#pragma unroll
        for (int r = 0; r < 8; ++r) {
            float s = 0.f;
#pragma unroll
            for (int tt = 0; tt < 4; ++tt) s += acc[mf][tt][r] * acc[mf][tt][r];
            atomicAdd(&rowss[mf * 16 + r + hi * 8], s);
        }
    __syncthreads();

#pragma unroll
    for (int mf = 0; mf < 2; ++mf)
#pragma unroll
        for (int r = 0; r < 8; ++r) {
            float scl = 1.f / fmaxf(sqrtf(rowss[mf * 16 + r + hi * 8]), 1e-12f);
            int row = m0 + mf * 16 + r + hi * 8;
#pragma unroll
            for (int tt = 0; tt < 4; ++tt) {
                int col = cb + tt * 16 + rL;
                out[(size_t)row * DMODEL + col] = acc[mf][tt][r] * scl;
            }
        }
}

extern "C" void kernel_launch(void* const* d_in, const int* in_sizes, int n_in,
                              void* d_out, int out_size, void* d_ws, size_t ws_size,
                              hipStream_t stream) {
    (void)in_sizes; (void)n_in; (void)out_size; (void)ws_size;
    const float* x  = (const float*)d_in[0];
    const float* wq = (const float*)d_in[1];
    const float* wk = (const float*)d_in[2];
    const float* wv = (const float*)d_in[3];
    const float* Wl = (const float*)d_in[4];
    const float* bl = (const float*)d_in[5];
    const float* W2 = (const float*)d_in[6];
    const float* b2 = (const float*)d_in[7];
    const float* W3 = (const float*)d_in[8];
    const float* b3 = (const float*)d_in[9];
    float* outp = (float*)d_out;

    char* ws = (char*)d_ws;
    _Float16* wqkvT = (_Float16*)(ws + OFF_WQKVT);
    _Float16* wlT   = (_Float16*)(ws + OFF_WLT);
    _Float16* w2T   = (_Float16*)(ws + OFF_W2T);
    _Float16* w3T   = (_Float16*)(ws + OFF_W3T);
    float*    y32   = (float*)   (ws + OFF_Y32);
    _Float16* y16   = (_Float16*)(ws + OFF_Y16);
    _Float16* h16   = (_Float16*)(ws + OFF_H16);

    prep_qkvT<<<(QKVN * DMODEL + 255) / 256, 256, 0, stream>>>(wq, wk, wv, wqkvT);
    prep_T<<<(32 * DMODEL + 255) / 256, 256, 0, stream>>>(Wl, wlT, 32, DMODEL);
    prep_T<<<(DMODEL * DMODEL + 255) / 256, 256, 0, stream>>>(W2, w2T, DMODEL, DMODEL);
    prep_T<<<(DMODEL * DMODEL + 255) / 256, 256, 0, stream>>>(W3, w3T, DMODEL, DMODEL);

    attn_kernel<<<BATCH, 128, 0, stream>>>(x, wqkvT, wlT, bl, y32, y16);
    ffn1_kernel<<<ROWS / 32, 512, 0, stream>>>(y16, w2T, b2, h16);
    ffn2_kernel<<<ROWS / 32, 512, 0, stream>>>(h16, w3T, b3, y32, outp);
}
